// EquivariantTransformer_90357521973982
// MI455X (gfx1250) — matrix operand order repaired and checked by reading, not compiled, not run on hardware
//
#include <hip/hip_runtime.h>
#include <hip/hip_bf16.h>

// ---------------------------------------------------------------------------
// Shapes (compile-time constants from the reference)
// ---------------------------------------------------------------------------
#define BB   4
#define NN   1024
#define MM   128
#define CC   512
#define HH   8
#define GG   3
#define KDIM 16
#define DHH  64
#define BN   (BB * NN)          // 4096 rows

typedef __attribute__((ext_vector_type(16))) __bf16 v16bf;
typedef __attribute__((ext_vector_type(8)))  float  v8f;

// ---------------------------------------------------------------------------
// Kernel 1: squared group-distance + top-M neighbor selection (bitonic sort).
// One workgroup (256 thr, 8 waves) per query row (b,n). Keys packed as
// (dist_bits<<32 | j) -> ascending sort gives nearest-first, ties by index,
// matching jax.lax.top_k(-dists) semantics.
// ---------------------------------------------------------------------------
__global__ __launch_bounds__(256) void topk_kernel(
    const float* __restrict__ pg, const unsigned char* __restrict__ mask,
    int* __restrict__ nbhd_idx, float* __restrict__ nbhd_g) {
  const int row = blockIdx.x;        // b*N + n
  const int bb  = row >> 10;
  const int tid = threadIdx.x;
  __shared__ unsigned long long keys[NN];

  for (int j = tid; j < NN; j += 256) {
    const float* g = pg + ((size_t)row * NN + j) * GG;
    float d2 = g[0] * g[0] + g[1] * g[1] + g[2] * g[2];
    if (!mask[bb * NN + j]) d2 = 3.0e38f;   // masked -> pushed to the end
    keys[j] = ((unsigned long long)__float_as_uint(d2) << 32) | (unsigned)j;
  }
  __syncthreads();

  for (int kk = 2; kk <= NN; kk <<= 1) {
    for (int j = kk >> 1; j > 0; j >>= 1) {
      for (int i = tid; i < NN; i += 256) {
        int ixj = i ^ j;
        if (ixj > i) {
          unsigned long long a = keys[i], b = keys[ixj];
          bool up = ((i & kk) == 0);
          if ((a > b) == up) { keys[i] = b; keys[ixj] = a; }
        }
      }
      __syncthreads();
    }
  }

  if (tid < MM) {
    int m = tid;
    int j = (int)(keys[m] & 0xffffffffu);
    nbhd_idx[(size_t)row * MM + m] = j;
    const float* g = pg + ((size_t)row * NN + j) * GG;
    float* o = nbhd_g + ((size_t)row * MM + m) * GG;
    o[0] = g[0]; o[1] = g[1]; o[2] = g[2];
  }
}

// ---------------------------------------------------------------------------
// f32 -> bf16 converts (plain + transpose for GEMM B operands)
// ---------------------------------------------------------------------------
__global__ __launch_bounds__(256) void cvt_bf16(
    const float* __restrict__ src, __bf16* __restrict__ dst, int n) {
  int i = blockIdx.x * 256 + threadIdx.x;
  if (i < n) dst[i] = (__bf16)src[i];
}

__global__ __launch_bounds__(256) void cvt_transpose_bf16(
    const float* __restrict__ W, __bf16* __restrict__ WT, int K, int Nc) {
  int i = blockIdx.x * 256 + threadIdx.x;
  if (i < K * Nc) {
    int k = i / Nc, n = i - k * Nc;
    WT[(size_t)n * K + k] = (__bf16)W[(size_t)k * Nc + n];
  }
}

// ---------------------------------------------------------------------------
// Kernel 2: bf16 WMMA GEMM  C = A(bf16, Mr x K) @ B (weights stored N x K,
// pre-transposed) + bias. 128 threads = 4 waves; each wave owns one 16x16 C
// tile, block covers a 16(row) x 64(col) macro-tile.
// Fragment layouts per CDNA5 ISA 7.12.2 (wave32):
//   A 16x32 bf16 : lane m = lane&15, VGPR v holds K pair
//                  k = 2v + (v>=4 ? 8 : 0) + (lane>=16 ? 8 : 0)
//   B 32x16 bf16 : lane n = lane&15, k = 2v + (v>=4 ? 8 : 0) + (lane>=16 ? 8 : 0)
//                  (weights pre-transposed so this is a contiguous u32 load)
//   C/D 16x16 f32: VGPR r -> row r + (lane>=16 ? 8 : 0), col = lane&15
// Output: f32 to Cf if non-null, else bf16 to Cb (for L2-traffic-halved k/v).
// ---------------------------------------------------------------------------
__global__ __launch_bounds__(128) void gemm_bf16_wmma(
    const __bf16* __restrict__ A, const __bf16* __restrict__ BT,
    const float* __restrict__ bias, float* __restrict__ Cf,
    __bf16* __restrict__ Cb, int Mrows, int K, int Nc) {
  const int lane = threadIdx.x & 31;
  const int wave = threadIdx.x >> 5;
  const int row0 = blockIdx.x * 16;
  const int col0 = blockIdx.y * 64 + wave * 16;
  const int mn   = lane & 15;
  const int hi   = lane >> 4;          // 0 or 1

  union U { v16bf v; unsigned u[8]; };
  v8f acc = {};

  for (int k0 = 0; k0 < K; k0 += 32) {
    U a, b;
#pragma unroll
    for (int v = 0; v < 8; ++v) {
      const int ka = 2 * v + (v >= 4 ? 8 : 0) + (hi ? 8 : 0);
      a.u[v] = *(const unsigned*)(A + (size_t)(row0 + mn) * K + k0 + ka);
      const int kb = 2 * v + (v >= 4 ? 8 : 0) + (hi ? 8 : 0);
      b.u[v] = *(const unsigned*)(BT + (size_t)(col0 + mn) * K + k0 + kb);
    }
    acc = __builtin_amdgcn_wmma_f32_16x16x32_bf16(
        /*neg_a=*/false, a.v, /*neg_b=*/false, b.v,
        /*c_mod=*/(short)0, acc, /*reuse_a=*/false, /*reuse_b=*/false);
  }

  const float bv = bias[col0 + mn];
  if (Cf) {
#pragma unroll
    for (int r = 0; r < 8; ++r) {
      const int mr = r + (hi ? 8 : 0);
      Cf[(size_t)(row0 + mr) * Nc + col0 + mn] = acc[r] + bv;
    }
  } else {
#pragma unroll
    for (int r = 0; r < 8; ++r) {
      const int mr = r + (hi ? 8 : 0);
      Cb[(size_t)(row0 + mr) * Nc + col0 + mn] = (__bf16)(acc[r] + bv);
    }
  }
}

// ---------------------------------------------------------------------------
// Kernel 3: per-query attention. One workgroup per (b,n):
//   - q row scaled by 1/sqrt(DH) into LDS (f32)
//   - location MLP (3->16->16->8, silu) -> initial logits
//   - gathered packed-bf16 K dot products (bit-shift unpack, no cvt)
//   - masked softmax over M per head
//   - gathered packed-bf16 V aggregation (1 u32 = 2 channels per thread)
// K/V stored bf16 -> gather traffic halved (1 GB -> 512 MB through L2).
// ---------------------------------------------------------------------------
__device__ __forceinline__ float silu_f(float x) {
  return x / (1.0f + __expf(-x));
}
__device__ __forceinline__ float bf_lo(unsigned u) { return __uint_as_float(u << 16); }
__device__ __forceinline__ float bf_hi(unsigned u) { return __uint_as_float(u & 0xffff0000u); }

__global__ __launch_bounds__(256) void attn_kernel(
    const float* __restrict__ qf, const __bf16* __restrict__ kbf,
    const __bf16* __restrict__ vbf, const int* __restrict__ nbhd_idx,
    const float* __restrict__ nbhd_g, const unsigned char* __restrict__ mask,
    const float* __restrict__ W1, const float* __restrict__ b1,
    const float* __restrict__ W2, const float* __restrict__ b2,
    const float* __restrict__ W3, const float* __restrict__ b3,
    float* __restrict__ ao) {
  const int row = blockIdx.x;          // b*N + n
  const int bb  = row >> 10;
  const int tid = threadIdx.x;
  const unsigned* __restrict__ kW = (const unsigned*)kbf;  // 2 bf16 per u32
  const unsigned* __restrict__ vW = (const unsigned*)vbf;

  __shared__ float qs[CC];
  __shared__ float sc[MM][HH];
  __shared__ int   sidx[MM];
  __shared__ unsigned char smask[MM];
  __shared__ float sinv[HH];

  for (int c = tid; c < CC; c += 256)
    qs[c] = qf[(size_t)row * CC + c] * 0.125f;   // 1/sqrt(64)

  if (tid < MM) {
    const int m = tid;
    const int j = nbhd_idx[(size_t)row * MM + m];
    sidx[m]  = j;
    smask[m] = mask[bb * NN + j];
    // warm L2/L0 for upcoming gathers of this neighbor's K/V rows
    __builtin_prefetch(kW + ((size_t)bb * NN + j) * (CC / 2), 0, 1);
    __builtin_prefetch(vW + ((size_t)bb * NN + j) * (CC / 2), 0, 1);

    // location MLP on lie-algebra offset
    float g0 = nbhd_g[((size_t)row * MM + m) * GG + 0];
    float g1 = nbhd_g[((size_t)row * MM + m) * GG + 1];
    float g2 = nbhd_g[((size_t)row * MM + m) * GG + 2];
    float h1[KDIM];
#pragma unroll
    for (int o = 0; o < KDIM; ++o)
      h1[o] = silu_f(b1[o] + g0 * W1[0 * KDIM + o] + g1 * W1[1 * KDIM + o] +
                     g2 * W1[2 * KDIM + o]);
    float h2[KDIM];
#pragma unroll
    for (int o = 0; o < KDIM; ++o) {
      float t = b2[o];
#pragma unroll
      for (int jj = 0; jj < KDIM; ++jj) t += h1[jj] * W2[jj * KDIM + o];
      h2[o] = silu_f(t);
    }
#pragma unroll
    for (int o = 0; o < HH; ++o) {
      float t = b3[o];
#pragma unroll
      for (int jj = 0; jj < KDIM; ++jj) t += h2[jj] * W3[jj * HH + o];
      sc[m][o] = t;
    }
  }
  __syncthreads();

  // multi-head dot products against gathered packed-bf16 keys
  for (int e = tid; e < MM * HH; e += 256) {
    const int m = e >> 3, h = e & 7;
    const unsigned* krow =
        kW + ((size_t)bb * NN + sidx[m]) * (CC / 2) + h * (DHH / 2);
    const float* qh = qs + h * DHH;
    float s = 0.0f;
#pragma unroll 8
    for (int i = 0; i < DHH / 2; ++i) {
      const unsigned u = krow[i];
      s += qh[2 * i] * bf_lo(u) + qh[2 * i + 1] * bf_hi(u);
    }
    sc[m][h] += s;
  }
  __syncthreads();

  // masked softmax over neighborhood axis, per head
  if (tid < HH) {
    const int h = tid;
    float mx = -3.0e38f;
    for (int m = 0; m < MM; ++m) {
      float x = smask[m] ? sc[m][h] : -1.0e38f;
      mx = (x > mx) ? x : mx;
    }
    float sum = 0.0f;
    for (int m = 0; m < MM; ++m) {
      float x = smask[m] ? sc[m][h] : -1.0e38f;
      float ex = __expf(x - mx);
      sc[m][h] = ex;
      sum += ex;
    }
    sinv[h] = 1.0f / sum;
  }
  __syncthreads();

  // weighted aggregate of gathered packed-bf16 values:
  // thread t owns u32 column t (channels 2t, 2t+1); per-m row reads coalesce.
  {
    const int cw = tid;                 // 0..255 u32 columns
    const int h  = cw >> 5;             // 32 u32 per head
    float acc0 = 0.0f, acc1 = 0.0f;
    for (int m = 0; m < MM; ++m) {
      const float w = sc[m][h];
      const unsigned u = vW[((size_t)bb * NN + sidx[m]) * (CC / 2) + cw];
      acc0 += w * bf_lo(u);
      acc1 += w * bf_hi(u);
    }
    const float inv = sinv[h];
    ao[(size_t)row * CC + 2 * cw]     = acc0 * inv;
    ao[(size_t)row * CC + 2 * cw + 1] = acc1 * inv;
  }
}

// ---------------------------------------------------------------------------
// Launch: topk -> converts -> q/k/v WMMA GEMMs -> attention -> out WMMA GEMM
// ---------------------------------------------------------------------------
extern "C" void kernel_launch(void* const* d_in, const int* in_sizes, int n_in,
                              void* d_out, int out_size, void* d_ws, size_t ws_size,
                              hipStream_t stream) {
  const float*         pg    = (const float*)d_in[0];          // (B,N,N,G)
  const float*         coset = (const float*)d_in[1];          // (B,N,C)
  const unsigned char* mask  = (const unsigned char*)d_in[2];  // (B,N) bool
  const float* W1 = (const float*)d_in[3];  const float* b1 = (const float*)d_in[4];
  const float* W2 = (const float*)d_in[5];  const float* b2 = (const float*)d_in[6];
  const float* W3 = (const float*)d_in[7];  const float* b3 = (const float*)d_in[8];
  const float* Wq = (const float*)d_in[9];  const float* bq = (const float*)d_in[10];
  const float* Wk = (const float*)d_in[11]; const float* bk = (const float*)d_in[12];
  const float* Wv = (const float*)d_in[13]; const float* bv = (const float*)d_in[14];
  const float* Wo = (const float*)d_in[15]; const float* bo = (const float*)d_in[16];
  float* out = (float*)d_out;

  // workspace carve-up (bytes), total ~44 MB
  char* ws = (char*)d_ws;
  int*    nbhd_idx = (int*)   (ws + 0);          //  2,097,152
  float*  nbhd_g   = (float*) (ws + 2097152);    //  6,291,456
  __bf16* Xbf      = (__bf16*)(ws + 8388608);    //  4,194,304
  __bf16* Wbf      = (__bf16*)(ws + 12582912);   //  2,097,152 (4 x 512x512)
  float*  qf       = (float*) (ws + 14680064);   //  8,388,608
  __bf16* kbf      = (__bf16*)(ws + 23068672);   //  4,194,304
  __bf16* vbf      = (__bf16*)(ws + 27262976);   //  4,194,304
  float*  ao       = (float*) (ws + 31457280);   //  8,388,608
  __bf16* aobf     = (__bf16*)(ws + 39845888);   //  4,194,304

  // 1) neighborhoods
  topk_kernel<<<BN, 256, 0, stream>>>(pg, mask, nbhd_idx, nbhd_g);

  // 2) bf16 operands
  const int nX = BN * CC;                         // 2M elements
  cvt_bf16<<<(nX + 255) / 256, 256, 0, stream>>>(coset, Xbf, nX);
  const int nW = CC * CC;                         // 256K elements per weight
  cvt_transpose_bf16<<<(nW + 255) / 256, 256, 0, stream>>>(Wq, Wbf + 0 * (size_t)nW, CC, CC);
  cvt_transpose_bf16<<<(nW + 255) / 256, 256, 0, stream>>>(Wk, Wbf + 1 * (size_t)nW, CC, CC);
  cvt_transpose_bf16<<<(nW + 255) / 256, 256, 0, stream>>>(Wv, Wbf + 2 * (size_t)nW, CC, CC);
  cvt_transpose_bf16<<<(nW + 255) / 256, 256, 0, stream>>>(Wo, Wbf + 3 * (size_t)nW, CC, CC);

  // 3) q/k/v projections (WMMA); k/v emitted as bf16 to halve gather traffic
  dim3 gg(BN / 16, CC / 64);
  gemm_bf16_wmma<<<gg, 128, 0, stream>>>(Xbf, Wbf + 0 * (size_t)nW, bq, qf,
                                         (__bf16*)nullptr, BN, CC, CC);
  gemm_bf16_wmma<<<gg, 128, 0, stream>>>(Xbf, Wbf + 1 * (size_t)nW, bk,
                                         (float*)nullptr, kbf, BN, CC, CC);
  gemm_bf16_wmma<<<gg, 128, 0, stream>>>(Xbf, Wbf + 2 * (size_t)nW, bv,
                                         (float*)nullptr, vbf, BN, CC, CC);

  // 4) attention core (gathers hit L2; bf16 k/v are 4MB each vs 192MB L2)
  attn_kernel<<<BN, 256, 0, stream>>>(qf, kbf, vbf, nbhd_idx, nbhd_g, mask,
                                      W1, b1, W2, b2, W3, b3, ao);

  // 5) output projection (WMMA) -> d_out (fp32)
  cvt_bf16<<<(nX + 255) / 256, 256, 0, stream>>>(ao, aobf, nX);
  gemm_bf16_wmma<<<gg, 128, 0, stream>>>(aobf, Wbf + 3 * (size_t)nW, bo, out,
                                         (__bf16*)nullptr, BN, CC, CC);
}
